// MixtralForCausalLM_50835232916128
// MI455X (gfx1250) — matrix operand order repaired and checked by reading, not compiled, not run on hardware
//
#include <hip/hip_runtime.h>
#include <hip/hip_bf16.h>

typedef __attribute__((ext_vector_type(16))) __bf16 v16bf;
typedef __attribute__((ext_vector_type(8)))  float  v8f;

#define T_TOK 4096
#define H_DIM 1024
#define E_NUM 8
#define I_DIM 3584
#define BM    128
#define BN    128
#define BK    32
#define MBLK  (T_TOK / BM)   // 32 token-blocks per expert (worst case)

union FragU { uint4 u[2]; v16bf v; };
union PackU { __bf16 b[2]; unsigned u; };
union Flt4U { float4 v; float f[4]; };

// ---------------------------------------------------------------------------
// 1) Router: logits -> top-2 -> renormalized weights -> per-expert lists
// ---------------------------------------------------------------------------
__global__ __launch_bounds__(256)
void router_kernel(const float* __restrict__ x, const float* __restrict__ gw,
                   int* __restrict__ counts, int* __restrict__ toks,
                   float* __restrict__ wts) {
  const int wave = threadIdx.x >> 5;
  const int lane = threadIdx.x & 31;
  const int t = blockIdx.x * 8 + wave;
  if (t >= T_TOK) return;

  float acc[E_NUM] = {};
  const float* xr = x + (size_t)t * H_DIM;
  for (int h = lane; h < H_DIM; h += 32) {
    const float xv = xr[h];
    const float* g = gw + h * E_NUM;
#pragma unroll
    for (int e = 0; e < E_NUM; ++e) acc[e] += xv * g[e];
  }
#pragma unroll
  for (int off = 16; off >= 1; off >>= 1) {
#pragma unroll
    for (int e = 0; e < E_NUM; ++e) acc[e] += __shfl_xor(acc[e], off, 32);
  }
  if (lane == 0) {
    int i0 = 0; float l0 = acc[0];
#pragma unroll
    for (int e = 1; e < E_NUM; ++e) if (acc[e] > l0) { l0 = acc[e]; i0 = e; }
    int i1 = -1; float l1 = -3.4e38f;
#pragma unroll
    for (int e = 0; e < E_NUM; ++e)
      if (e != i0 && acc[e] > l1) { l1 = acc[e]; i1 = e; }
    // renormalized top-2 softmax weights: full-softmax denominator cancels
    const float r  = __expf(l1 - l0);
    const float w0 = 1.0f / (1.0f + r);
    const float w1 = 1.0f - w0;
    const int p0 = atomicAdd(&counts[i0], 1);
    toks[i0 * T_TOK + p0] = (t << 1);
    wts [i0 * T_TOK + p0] = w0;
    const int p1 = atomicAdd(&counts[i1], 1);
    toks[i1 * T_TOK + p1] = (t << 1) | 1;
    wts [i1 * T_TOK + p1] = w1;
  }
}

// ---------------------------------------------------------------------------
// 2) hidden fp32 -> bf16
// ---------------------------------------------------------------------------
__global__ __launch_bounds__(256)
void cvt_hidden_kernel(const float* __restrict__ x, __bf16* __restrict__ xb) {
  const int i = blockIdx.x * blockDim.x + threadIdx.x;   // over T*H/4
  const float4 v = ((const float4*)x)[i];
  union { __bf16 b[4]; uint2 u; } o;
  o.b[0] = (__bf16)v.x; o.b[1] = (__bf16)v.y;
  o.b[2] = (__bf16)v.z; o.b[3] = (__bf16)v.w;
  ((uint2*)xb)[i] = o.u;
}

// ---------------------------------------------------------------------------
// Shared GEMM skeleton: 128x128x32 workgroup tile, 4x2 wave grid (wave tile
// 32Mx64N = 2 A-frags x 4 B-frags -> 8 accumulators), double-buffered LDS with
// a single barrier per K-step, XOR 16B-chunk address swizzle, fp32->bf16
// conversion fused into B staging. All register-array indexing is static
// (dynamic indices force scratch spills); B stores accept a 4-way LDS bank
// conflict, which stays hidden under the 8 WMMAs, while fragment reads remain
// conflict-free via the chunk swizzle.
// ---------------------------------------------------------------------------
template <typename EpiFn>
__device__ __forceinline__
void gemm_body(const __bf16* __restrict__ asrc, const float* __restrict__ bsrc,
               const int kdim, const size_t ldb,
               __bf16 (&At)[2][BM * BK], __bf16 (&Bt)[2][BN * BK],
               const int row0, const int n_e, EpiFn&& epi) {
  const int tid  = threadIdx.x;
  const int wave = tid >> 5;
  const int lane = tid & 31;
  // A staging: thread -> (row, 16-elem half)
  const int arow  = tid >> 1;
  const int ahalf = tid & 1;
  const int aswz  = (arow >> 2) & 3;
  const int apc0  = (2 * ahalf)     ^ aswz;
  const int apc1  = (2 * ahalf + 1) ^ aswz;
  // B staging: lane -> (k-pair, two 4-col groups at n and n+64)
  const int bcol = lane & 15;                 // col group: n = 4*bcol (+64)
  const int kp   = 2 * wave + (lane >> 4);    // k-pair: k = 2*kp
  const int bpc  = (kp >> 2) ^ (bcol & 3);    // physical chunk for stores
  const int bwrd = kp & 3;
  // fragment coords: wave (wm, wn) covers rows wm*32..+32, cols wn*64..+64
  const int wm   = wave & 3;
  const int wn   = wave >> 2;
  const int kq   = lane >> 4;
  const int fm   = lane & 15;
  const int fswz = (fm >> 2) & 3;
  const int c0   = (2 * kq) ^ fswz;

  v8f acc[2][4] = {};
  uint4 pa0, pa1;
  Flt4U pb[2][2];                       // [k-row][col-group]
  // prologue: fetch tile 0 into registers
  pa0        = ((const uint4*)(asrc))[0];
  pa1        = ((const uint4*)(asrc))[1];
  pb[0][0].v = ((const float4*)(bsrc))[0];
  pb[0][1].v = ((const float4*)(bsrc + 64))[0];
  pb[1][0].v = ((const float4*)(bsrc + ldb))[0];
  pb[1][1].v = ((const float4*)(bsrc + ldb + 64))[0];

  const int NK = kdim / BK;
  for (int it = 0; it < NK; ++it) {
    const int cur = it & 1;
    { // store prefetched tile into LDS[cur]  (all indices compile-time)
      __bf16* Ab = &At[cur][0];
      *(uint4*)&Ab[arow * BK + apc0 * 8] = pa0;
      *(uint4*)&Ab[arow * BK + apc1 * 8] = pa1;
      __bf16* Bb = &Bt[cur][0];
#pragma unroll
      for (int g2 = 0; g2 < 2; ++g2) {
        const int nbase = 4 * bcol + g2 * 64;
#pragma unroll
        for (int j = 0; j < 4; ++j) {
          PackU p;
          p.b[0] = (__bf16)pb[0][g2].f[j];
          p.b[1] = (__bf16)pb[1][g2].f[j];
          *(unsigned*)&Bb[(nbase + j) * BK + bpc * 8 + bwrd * 2] = p.u;
        }
      }
    }
    __syncthreads();   // single barrier per K-step (double-buffer safe)
    if (it + 1 < NK) { // prefetch next tile from global (overlaps WMMAs below)
      const int k0 = (it + 1) * BK;
      pa0        = ((const uint4*)(asrc + k0))[0];
      pa1        = ((const uint4*)(asrc + k0))[1];
      const float* bk0 = bsrc + (size_t)k0 * ldb;
      pb[0][0].v = ((const float4*)(bk0))[0];
      pb[0][1].v = ((const float4*)(bk0 + 64))[0];
      pb[1][0].v = ((const float4*)(bk0 + ldb))[0];
      pb[1][1].v = ((const float4*)(bk0 + ldb + 64))[0];
    }
    { // compute on LDS[cur]; B frags consumed in two pipelined pairs
      const __bf16* Ab = &At[cur][0];
      const __bf16* Bb = &Bt[cur][0];
      FragU a[2];
#pragma unroll
      for (int mi = 0; mi < 2; ++mi) {
        const int mrow = wm * 32 + mi * 16 + fm;
        const int mswz = (mrow >> 2) & 3;
        a[mi].u[0] = *(const uint4*)&Ab[mrow * BK + ((2 * kq) ^ mswz) * 8];
        a[mi].u[1] = *(const uint4*)&Ab[mrow * BK + ((2 * kq + 1) ^ mswz) * 8];
      }
#pragma unroll
      for (int half = 0; half < 2; ++half) {
        FragU b0, b1;
        const int nA = wn * 64 + (half * 2 + 0) * 16 + fm;
        const int nB = wn * 64 + (half * 2 + 1) * 16 + fm;
        b0.u[0] = *(const uint4*)&Bb[nA * BK + c0 * 8];
        b0.u[1] = *(const uint4*)&Bb[nA * BK + (c0 ^ 1) * 8];
        b1.u[0] = *(const uint4*)&Bb[nB * BK + c0 * 8];
        b1.u[1] = *(const uint4*)&Bb[nB * BK + (c0 ^ 1) * 8];
#pragma unroll
        for (int mi = 0; mi < 2; ++mi)
          acc[mi][half * 2 + 0] = __builtin_amdgcn_wmma_f32_16x16x32_bf16(
              false, a[mi].v, false, b0.v, (short)0, acc[mi][half * 2 + 0],
              false, false);
#pragma unroll
        for (int mi = 0; mi < 2; ++mi)
          acc[mi][half * 2 + 1] = __builtin_amdgcn_wmma_f32_16x16x32_bf16(
              false, a[mi].v, false, b1.v, (short)0, acc[mi][half * 2 + 1],
              false, false);
      }
    }
  }

  // epilogue: col = n0 + cbase + f*16 inside epi
  const int cn    = lane & 15;
  const int cbase = wn * 64 + cn;
#pragma unroll
  for (int mi = 0; mi < 2; ++mi) {
#pragma unroll
    for (int v = 0; v < 8; ++v) {
      const int rloc = wm * 32 + mi * 16 + v + kq * 8;
      if (row0 + rloc < n_e) epi(acc[mi], v, rloc, cbase);
    }
  }
}

// ---------------------------------------------------------------------------
// 3) GEMM1: act[r, I] = silu( X[tok(r), :] @ w1[e] )   (bf16 in, bf16 out)
// ---------------------------------------------------------------------------
__global__ __launch_bounds__(256, 2)
void moe_gemm1_kernel(const __bf16* __restrict__ xb, const float* __restrict__ w1,
                      const int* __restrict__ counts, const int* __restrict__ toks,
                      __bf16* __restrict__ act) {
  const int e    = blockIdx.x / MBLK;
  const int mb   = blockIdx.x % MBLK;
  const int n_e  = counts[e];
  const int row0 = mb * BM;
  if (row0 >= n_e) return;
  const int n0 = blockIdx.y * BN;

  __shared__ __bf16 At[2][BM * BK];
  __shared__ __bf16 Bt[2][BN * BK];
  __shared__ int    tokS[BM];

  if (threadIdx.x < BM) {
    const int r = row0 + threadIdx.x;
    tokS[threadIdx.x] = toks[e * T_TOK + (r < n_e ? r : n_e - 1)];
  }
  __syncthreads();

  const __bf16* asrc =
      xb + (size_t)(tokS[threadIdx.x >> 1] >> 1) * H_DIM + (threadIdx.x & 1) * 16;
  const int kp_ = 2 * (threadIdx.x >> 5) + ((threadIdx.x & 31) >> 4);
  const int bc_ = threadIdx.x & 15;
  const float* bsrc =
      w1 + ((size_t)e * H_DIM + 2 * kp_) * (size_t)I_DIM + n0 + 4 * bc_;

  gemm_body(asrc, bsrc, H_DIM, (size_t)I_DIM, At, Bt, row0, n_e,
            [&](const v8f (&accm)[4], int v, int rloc, int cbase) {
              const int dest = tokS[rloc];          // packed (t<<1)|slot
#pragma unroll
              for (int f = 0; f < 4; ++f) {
                const float xv = accm[f][v];
                const float sg = __builtin_amdgcn_rcpf(1.0f + __expf(-xv));
                act[(size_t)dest * I_DIM + n0 + cbase + f * 16] =
                    (__bf16)(xv * sg);
              }
            });
}

// ---------------------------------------------------------------------------
// 4) GEMM2: partial[r, H] = w(t,e) * ( act[r, :] @ w2[e] )   (fp32 out)
// ---------------------------------------------------------------------------
__global__ __launch_bounds__(256, 2)
void moe_gemm2_kernel(const __bf16* __restrict__ act, const float* __restrict__ w2,
                      const int* __restrict__ counts, const int* __restrict__ toks,
                      const float* __restrict__ wts, float* __restrict__ partial) {
  const int e    = blockIdx.x / MBLK;
  const int mb   = blockIdx.x % MBLK;
  const int n_e  = counts[e];
  const int row0 = mb * BM;
  if (row0 >= n_e) return;
  const int n0 = blockIdx.y * BN;

  __shared__ __bf16 At[2][BM * BK];
  __shared__ __bf16 Bt[2][BN * BK];
  __shared__ int    tokS[BM];
  __shared__ float  wtS[BM];

  if (threadIdx.x < BM) {
    const int r  = row0 + threadIdx.x;
    const int rc = (r < n_e ? r : n_e - 1);
    tokS[threadIdx.x] = toks[e * T_TOK + rc];
    wtS [threadIdx.x] = wts [e * T_TOK + rc];
  }
  __syncthreads();

  const __bf16* asrc =
      act + (size_t)tokS[threadIdx.x >> 1] * I_DIM + (threadIdx.x & 1) * 16;
  const int kp_ = 2 * (threadIdx.x >> 5) + ((threadIdx.x & 31) >> 4);
  const int bc_ = threadIdx.x & 15;
  const float* bsrc =
      w2 + ((size_t)e * I_DIM + 2 * kp_) * (size_t)H_DIM + n0 + 4 * bc_;

  gemm_body(asrc, bsrc, I_DIM, (size_t)H_DIM, At, Bt, row0, n_e,
            [&](const v8f (&accm)[4], int v, int rloc, int cbase) {
              const int   dest = tokS[rloc];
              const float sc   = wtS[rloc];
#pragma unroll
              for (int f = 0; f < 4; ++f)
                partial[(size_t)dest * H_DIM + n0 + cbase + f * 16] =
                    sc * accm[f][v];
            });
}

// ---------------------------------------------------------------------------
// 5) Combine the two expert-slot partials per token (deterministic)
// ---------------------------------------------------------------------------
__global__ __launch_bounds__(256)
void combine_kernel(const float* __restrict__ partial, float* __restrict__ out) {
  const int i  = blockIdx.x * blockDim.x + threadIdx.x;  // over T*H/4
  const int hq = H_DIM / 4;
  const int t  = i / hq;
  const int h4 = i % hq;
  const float4 a = ((const float4*)(partial + (size_t)(2 * t)     * H_DIM))[h4];
  const float4 b = ((const float4*)(partial + (size_t)(2 * t + 1) * H_DIM))[h4];
  float4 o; o.x = a.x + b.x; o.y = a.y + b.y; o.z = a.z + b.z; o.w = a.w + b.w;
  ((float4*)out)[i] = o;
}

// ---------------------------------------------------------------------------
extern "C" void kernel_launch(void* const* d_in, const int* in_sizes, int n_in,
                              void* d_out, int out_size, void* d_ws, size_t ws_size,
                              hipStream_t stream) {
  const float* hidden = (const float*)d_in[0];
  const float* gate_w = (const float*)d_in[1];
  const float* w1     = (const float*)d_in[2];
  const float* w2     = (const float*)d_in[3];
  float* out = (float*)d_out;

  char* ws = (char*)d_ws;
  const size_t off_xb      = 0;
  const size_t off_act     = off_xb   + (size_t)T_TOK * H_DIM * 2;          //  8.4 MB
  const size_t off_partial = off_act  + (size_t)2 * T_TOK * I_DIM * 2;      // 58.7 MB
  const size_t off_counts  = off_partial + (size_t)2 * T_TOK * H_DIM * 4;   // 33.6 MB
  const size_t off_toks    = off_counts + 256;
  const size_t off_wts     = off_toks + (size_t)E_NUM * T_TOK * 4;

  __bf16* xb      = (__bf16*)(ws + off_xb);
  __bf16* act     = (__bf16*)(ws + off_act);
  float*  partial = (float*)(ws + off_partial);
  int*    counts  = (int*)  (ws + off_counts);
  int*    toks    = (int*)  (ws + off_toks);
  float*  wts     = (float*)(ws + off_wts);

  hipMemsetAsync(counts, 0, E_NUM * sizeof(int), stream);

  router_kernel<<<T_TOK / 8, 256, 0, stream>>>(hidden, gate_w, counts, toks, wts);
  cvt_hidden_kernel<<<(T_TOK * H_DIM / 4) / 256, 256, 0, stream>>>(hidden, xb);
  moe_gemm1_kernel<<<dim3(E_NUM * MBLK, I_DIM / BN), 256, 0, stream>>>(
      xb, w1, counts, toks, act);
  moe_gemm2_kernel<<<dim3(E_NUM * MBLK, H_DIM / BN), 256, 0, stream>>>(
      act, w2, counts, toks, wts, partial);
  combine_kernel<<<(T_TOK * H_DIM / 4) / 256, 256, 0, stream>>>(partial, out);
}
